// GNN_40613210751535
// MI455X (gfx1250) — matrix operand order repaired and checked by reading, not compiled, not run on hardware
//
#include <hip/hip_runtime.h>
#include <stdint.h>

// ---------------- CDNA5 WMMA types ----------------
typedef __bf16 bf16_t;
typedef bf16_t v16bf __attribute__((ext_vector_type(16)));
typedef float  v8f   __attribute__((ext_vector_type(8)));

#define N_NODES  50000
#define N_EDGES  600000
#define N_GRAPHS 64

__device__ __forceinline__ unsigned short f32_to_bf16(float f) {
  union { float f; unsigned int u; } a; a.f = f;
  unsigned int u = a.u;
  u += 0x7FFFu + ((u >> 16) & 1u);   // round-to-nearest-even
  return (unsigned short)(u >> 16);
}

// ---------------- utility ----------------
__global__ void zero_f32(float* __restrict__ p, int n) {
  int i = blockIdx.x * blockDim.x + threadIdx.x;
  if (i < n) p[i] = 0.0f;
}

// ---------------- degree ----------------
__global__ void deg_kernel(const long long* __restrict__ ei, float* __restrict__ deg) {
  int e = blockIdx.x * blockDim.x + threadIdx.x;
  if (e < N_EDGES) atomicAdd(&deg[(int)ei[N_EDGES + e]], 1.0f);
}

__global__ void invdeg_kernel(const float* __restrict__ deg, float* __restrict__ inv_deg) {
  int i = blockIdx.x * blockDim.x + threadIdx.x;
  if (i < N_NODES) inv_deg[i] = 1.0f / fmaxf(deg[i], 1.0f);
}

// ---------------- edge scatter: agg[dst] += feat[src]  (float4 per thread) ----
// s = log2(F/4); blockDim.x = 256 -> edges-per-block = 256 >> s
__global__ void scatter_kernel(const float* __restrict__ feat,
                               const long long* __restrict__ ei,
                               float* __restrict__ agg,
                               int F, int s) {
  const int tpe = 1 << s;
  const int fq = threadIdx.x & (tpe - 1);
  const int eLocal = threadIdx.x >> s;
  const int epb = blockDim.x >> s;
  const long long e = (long long)blockIdx.x * epb + eLocal;
  if (e >= N_EDGES) return;
  const int src = (int)ei[e];
  const int dst = (int)ei[N_EDGES + e];
  const float4 v = *(const float4*)(feat + (size_t)src * F + (fq << 2));
  float* ap = agg + (size_t)dst * F + (fq << 2);
  atomicAdd(ap + 0, v.x);
  atomicAdd(ap + 1, v.y);
  atomicAdd(ap + 2, v.z);
  atomicAdd(ap + 3, v.w);
}

// ---------------- pack A = [agg*inv_deg | x] as bf16, row length 2F ----------
// launch: grid = N_NODES blocks, blockDim = F
__global__ void pack_A(const float* __restrict__ agg,
                       const float* __restrict__ x,
                       const float* __restrict__ inv_deg,
                       unsigned short* __restrict__ P, int F) {
  const int n = blockIdx.x;
  const int f = threadIdx.x;
  const float inv = inv_deg[n];
  const size_t pbase = (size_t)n * ((size_t)F << 1);
  P[pbase + f]     = f32_to_bf16(agg[(size_t)n * F + f] * inv);
  P[pbase + F + f] = f32_to_bf16(x[(size_t)n * F + f]);
}

// ---------------- pack stacked weights [Wl;Wr] into WMMA B-fragment order ----
// For kt (K/32 tiles), nt (Fout/16 tiles), lane L (0..31), elem i (0..15):
//   linear index t = ((kt*nTiles + nt)*32 + L)*16 + i
//   K  = kt*32 + (L>=16 ? 8 : 0) + (i&7) + (i>=8 ? 16 : 0) ;  N = nt*16 + (L&15)
// (16-bit B 32x16 VGPR layout: lanes 0-15 hold K=0..7,16..23 packed 2/VGPR,
//  lanes 16-31 hold K=8..15,24..31)
__global__ void pack_W(const float* __restrict__ Wl,
                       const float* __restrict__ Wr,
                       unsigned short* __restrict__ Bp,
                       int Fin, int Fout) {
  const int total = (Fin << 1) * Fout;
  const int t = blockIdx.x * blockDim.x + threadIdx.x;
  if (t >= total) return;
  const int i = t & 15;
  const int L = (t >> 4) & 31;
  const int r = t >> 9;
  const int nTiles = Fout >> 4;
  const int kt = r / nTiles;
  const int nt = r - kt * nTiles;
  const int K = (kt << 5) + ((L >> 4) << 3) + (i & 7) + ((i >> 3) << 4);
  const int col = (nt << 4) + (L & 15);
  const float v = (K < Fin) ? Wl[(size_t)K * Fout + col]
                            : Wr[(size_t)(K - Fin) * Fout + col];
  Bp[t] = f32_to_bf16(v);
}

// ---------------- fused GEMM + bias + ReLU via v_wmma_f32_16x16x32_bf16 ------
// One wave computes a 16x32 output tile: one shared A fragment, TWO B
// fragments, TWO independent f32 accumulators -> the two WMMAs per K-step
// have no D->A/B RAW dependency, filling the bf16 WMMA hazard slots with
// real work instead of v_nops, and halving A-fragment load traffic.
// A frag per lane (16-bit A 16x32 layout): M = lane&15,
//   pairs at K = (lane>=16?8:0)+{0,2,4,6} and +16 -> two b128 loads/step.
// B frags: prepacked contiguous 32B/lane -> two b128 loads each per step.
__global__ void sage_gemm_wmma(const unsigned short* __restrict__ P,   // [N, K] bf16
                               const unsigned short* __restrict__ Bp,  // packed frags
                               const float* __restrict__ bias,         // [Fout]
                               float* __restrict__ H,                  // [N, Fout]
                               int K, int Fout) {
  const int lane = threadIdx.x & 31;
  const int wave = (blockIdx.x * blockDim.x + threadIdx.x) >> 5;
  const int colTiles = Fout >> 4;
  const int colPairs = colTiles >> 1;         // Fout in {256,128,64} -> even
  const int rowTiles = N_NODES >> 4;          // 3125, exact
  if (wave >= rowTiles * colPairs) return;    // wave-uniform exit
  const int rt = wave / colPairs;
  const int cp = wave - rt * colPairs;
  const int ct0 = cp << 1;                    // two adjacent 16-col tiles
  const int rowBase = rt << 4;
  const int m = rowBase + (lane & 15);
  const int hiK = (lane >> 4) << 3;           // 0 or 8

  v8f c0 = {0.f, 0.f, 0.f, 0.f, 0.f, 0.f, 0.f, 0.f};
  v8f c1 = {0.f, 0.f, 0.f, 0.f, 0.f, 0.f, 0.f, 0.f};

  // A: pointer walks K in steps of 32 bf16 = 16 dwords
  const unsigned int* aptr =
      (const unsigned int*)(P + (size_t)m * K) + (hiK >> 1);
  // B: uint4 units; frag (kt,ct) at ((kt*colTiles+ct)*32+lane)*2 uint4s
  const uint4* bptr = (const uint4*)Bp + (((size_t)ct0 * 32 + lane) << 1);
  const size_t bstep = (size_t)colTiles << 6;  // colTiles*32*2 uint4 per kt
  const int kSteps = K >> 5;

  for (int kt = 0; kt < kSteps; ++kt) {
    union { v16bf v; uint4 q[2]; } a;
    a.q[0] = *(const uint4*)(aptr);       // K pairs {hiK+0..7}
    a.q[1] = *(const uint4*)(aptr + 8);   // K pairs {16+hiK+0..7}
    union { v16bf v; uint4 q[2]; } b0, b1;
    b0.q[0] = bptr[0];
    b0.q[1] = bptr[1];
    b1.q[0] = bptr[64];                   // next 16-col tile: +32 lanes*2 uint4
    b1.q[1] = bptr[65];
    c0 = __builtin_amdgcn_wmma_f32_16x16x32_bf16(
        false, a.v, false, b0.v, (short)0, c0, false, false);
    c1 = __builtin_amdgcn_wmma_f32_16x16x32_bf16(
        false, a.v, false, b1.v, (short)0, c1, false, false);
    aptr += 16;
    bptr += bstep;
  }

  const int col0 = (ct0 << 4) + (lane & 15);
  const float bn0 = bias[col0];
  const float bn1 = bias[col0 + 16];
  const int rbase = rowBase + ((lane >> 4) << 3);   // C/D: VGPR r -> M=r or r+8
#pragma unroll
  for (int r = 0; r < 8; ++r) {
    float v0 = c0[r] + bn0;
    float v1 = c1[r] + bn1;
    v0 = v0 > 0.f ? v0 : 0.f;
    v1 = v1 > 0.f ? v1 : 0.f;
    float* hrow = H + (size_t)(rbase + r) * Fout + col0;
    hrow[0]  = v0;
    hrow[16] = v1;
  }
}

// ---------------- graph pooling + head ----------------
__global__ void pool_sum(const float* __restrict__ h3,
                         const long long* __restrict__ batch,
                         float* __restrict__ gsum) {
  const int n = blockIdx.x;     // grid = N_NODES, block = 64
  const int f = threadIdx.x;
  atomicAdd(&gsum[(int)batch[n] * 64 + f], h3[(size_t)n * 64 + f]);
}

__global__ void pool_cnt(const long long* __restrict__ batch, float* __restrict__ gcnt) {
  int n = blockIdx.x * blockDim.x + threadIdx.x;
  if (n < N_NODES) atomicAdd(&gcnt[(int)batch[n]], 1.0f);
}

__global__ void head_kernel(const float* __restrict__ gsum,
                            const float* __restrict__ gcnt,
                            const float* __restrict__ wlin,
                            const float* __restrict__ blin,
                            float* __restrict__ out) {
  const int g = threadIdx.x;
  if (g >= N_GRAPHS) return;
  const float inv = 1.0f / fmaxf(gcnt[g], 1.0f);
  float l0 = blin[0], l1 = blin[1];
#pragma unroll 8
  for (int f = 0; f < 64; ++f) {
    const float mv = gsum[g * 64 + f] * inv;
    l0 += mv * wlin[f * 2 + 0];
    l1 += mv * wlin[f * 2 + 1];
  }
  const float mx = fmaxf(l0, l1);
  const float lse = mx + logf(expf(l0 - mx) + expf(l1 - mx));
  out[g * 2 + 0] = l0 - lse;
  out[g * 2 + 1] = l1 - lse;
}

// ---------------- host orchestration ----------------
extern "C" void kernel_launch(void* const* d_in, const int* in_sizes, int n_in,
                              void* d_out, int out_size, void* d_ws, size_t ws_size,
                              hipStream_t stream) {
  (void)in_sizes; (void)n_in; (void)out_size; (void)ws_size;

  const float*     x     = (const float*)d_in[0];
  const long long* ei    = (const long long*)d_in[1];
  const long long* batch = (const long long*)d_in[2];
  const float* w1l = (const float*)d_in[3];
  const float* b1  = (const float*)d_in[4];
  const float* w1r = (const float*)d_in[5];
  const float* w2l = (const float*)d_in[6];
  const float* b2  = (const float*)d_in[7];
  const float* w2r = (const float*)d_in[8];
  const float* w3l = (const float*)d_in[9];
  const float* b3  = (const float*)d_in[10];
  const float* w3r = (const float*)d_in[11];
  const float* wlin = (const float*)d_in[12];
  const float* blin = (const float*)d_in[13];
  float* out = (float*)d_out;

  // workspace carve-up (256B aligned slices)
  char* wsb = (char*)d_ws;
  size_t off = 0;
  auto carve = [&](size_t bytes) -> char* {
    char* p = wsb + off;
    off = (off + bytes + 255) & ~(size_t)255;
    return p;
  };
  float* deg     = (float*)carve((size_t)N_NODES * 4);
  float* inv_deg = (float*)carve((size_t)N_NODES * 4);
  float* agg     = (float*)carve((size_t)N_NODES * 256 * 4);             // max F_in
  unsigned short* P = (unsigned short*)carve((size_t)N_NODES * 512 * 2); // max K
  float* h1      = (float*)carve((size_t)N_NODES * 256 * 4);
  float* h2      = (float*)carve((size_t)N_NODES * 128 * 4);
  float* h3      = (float*)carve((size_t)N_NODES * 64 * 4);
  unsigned short* Bp = (unsigned short*)carve((size_t)65536 * 2);        // max 2*Fin*Fout
  float* gsum    = (float*)carve((size_t)N_GRAPHS * 64 * 4);
  float* gcnt    = (float*)carve((size_t)N_GRAPHS * 4);

  auto zero = [&](float* p, int n) {
    zero_f32<<<(n + 255) / 256, 256, 0, stream>>>(p, n);
  };

  // degree / inverse degree
  zero(deg, N_NODES);
  deg_kernel<<<(N_EDGES + 255) / 256, 256, 0, stream>>>(ei, deg);
  invdeg_kernel<<<(N_NODES + 255) / 256, 256, 0, stream>>>(deg, inv_deg);

  // one SAGE layer: feat[N,Fin] -> hout[N,Fout]
  auto sage_layer = [&](const float* feat, const float* Wl, const float* bias,
                        const float* Wr, float* hout, int Fin, int Fout) {
    const int s = (Fin == 256) ? 6 : 5;            // log2(Fin/4), Fin in {128,256}
    const int epb = 256 >> s;
    zero(agg, N_NODES * Fin);
    scatter_kernel<<<(N_EDGES + epb - 1) / epb, 256, 0, stream>>>(feat, ei, agg, Fin, s);
    pack_A<<<N_NODES, Fin, 0, stream>>>(agg, feat, inv_deg, P, Fin);
    const int wtot = (Fin * 2) * Fout;
    pack_W<<<(wtot + 255) / 256, 256, 0, stream>>>(Wl, Wr, Bp, Fin, Fout);
    const int K = Fin * 2;
    const int tiles = (N_NODES / 16) * (Fout / 32);   // 16x32 tiles per wave
    const int threads = tiles * 32;
    sage_gemm_wmma<<<(threads + 255) / 256, 256, 0, stream>>>(P, Bp, bias, hout, K, Fout);
  };

  sage_layer(x,  w1l, b1, w1r, h1, 128, 256);
  sage_layer(h1, w2l, b2, w2r, h2, 256, 128);
  sage_layer(h2, w3l, b3, w3r, h3, 128, 64);

  // global mean pool + linear head + log_softmax
  zero(gsum, N_GRAPHS * 64);
  zero(gcnt, N_GRAPHS);
  pool_sum<<<N_NODES, 64, 0, stream>>>(h3, batch, gsum);
  pool_cnt<<<(N_NODES + 255) / 256, 256, 0, stream>>>(batch, gcnt);
  head_kernel<<<1, 64, 0, stream>>>(gsum, gcnt, wlin, blin, out);
}
